// DifferentiableInference_91010357002263
// MI455X (gfx1250) — matrix operand order repaired and checked by reading, not compiled, not run on hardware
//
#include <hip/hip_runtime.h>
#include <hip/hip_bf16.h>

// Problem constants (from reference setup_inputs)
#define B_DIM 32
#define C_DIM 128
#define G_DIM 4096
#define M_DIM 5

#define USE_TDM 1   // stage LDS table with tensor_load_to_lds (TDM); 0 = vmem copy loop

typedef __attribute__((ext_vector_type(16))) __bf16 v16bf;
typedef __attribute__((ext_vector_type(8)))  float  v8f;
typedef __attribute__((ext_vector_type(4)))  unsigned u32x4;
typedef __attribute__((ext_vector_type(8)))  int      i32x8;
typedef __attribute__((ext_vector_type(4)))  int      i32x4;

union BfOp  { unsigned u[8];  v16bf v; };
union AccOp { float    f[8];  v8f   v; };

// native bf16 converts -> v_cvt_pk_bf16_f32 (RNE) instead of manual rounding
__device__ __forceinline__ unsigned pack_bf16(float lo, float hi) {
  union { __bf16 b[2]; unsigned u; } t;
  t.b[0] = (__bf16)lo; t.b[1] = (__bf16)hi;
  return t.u;
}
__device__ __forceinline__ unsigned short to_bf16u(float f) {
  union { __bf16 b; unsigned short s; } t; t.b = (__bf16)f; return t.s;
}
__device__ __forceinline__ float bf2f(unsigned short h) {
  return __uint_as_float(((unsigned)h) << 16);   // exact widening
}

// ---------------------------------------------------------------------------
// Transpose v0 (B,G) f32 -> vt (G,B) bf16 so lane==b gathers are coalesced.
// ---------------------------------------------------------------------------
__global__ void k_transpose_in(const float* __restrict__ v0,
                               unsigned short* __restrict__ vt) {
  int t = blockIdx.x * blockDim.x + threadIdx.x;
  if (t >= B_DIM * G_DIM) return;
  int g = t >> 5, b = t & 31;
  vt[(size_t)g * B_DIM + b] = to_bf16u(v0[(size_t)b * G_DIM + g]);
}

// ---------------------------------------------------------------------------
// One T-step. Whole bf16 table (256 KB) lives in LDS; 32 waves, 1 g per wave;
// einsum via v_wmma_f32_16x16x32_bf16 (D(m,b) = Wmat-chunk x cv^T-chunk).
// ---------------------------------------------------------------------------
__global__ __launch_bounds__(1024, 1) void k_step(
    const unsigned short* __restrict__ vt_in,   // (G,B) bf16
    const long long*      __restrict__ X,       // (C,G,BODY) int64
    const unsigned char*  __restrict__ fmask,   // (C,G) bool
    const float*          __restrict__ W,       // (M,C) f32
    unsigned short*       __restrict__ vt_out,  // (G,B) bf16
    float*                __restrict__ out_f32) // (B,G) f32 or nullptr
{
  __shared__ unsigned short vt_lds[G_DIM * B_DIM];  // 256 KB of the 320 KB WGP LDS

  const int tid  = threadIdx.x;
  const int lane = tid & 31;
  const int wid  = tid >> 5;
  const int m    = lane & 15;   // WMMA A row (both lane halves map M = L%16)
  const int half = lane >> 4;

#if USE_TDM
  // ---- stage table with one Tensor Data Mover DMA (wave 0 issues) --------
  // D# (CDNA5 ISA): 2D tensor, data_size=8B, one row of 32768 elements
  // == 256 KB, tile == tensor, linear LDS write. Groups 2/3 unused (zero).
  if (wid == 0) {
    unsigned lds_base = (unsigned)(size_t)vt_lds;           // LDS byte offset
    unsigned long long ga = (unsigned long long)(size_t)vt_in;
    u32x4 g0;
    g0[0] = 1u;                                             // count=1, user D#
    g0[1] = lds_base;                                       // lds_addr
    g0[2] = (unsigned)ga;                                   // global_addr[31:0]
    g0[3] = (unsigned)((ga >> 32) & 0x1FFFFFFu) | (2u << 30); // addr[56:32]|type=2
    i32x8 g1;
    g1[0] = 0x30000;                  // wg_mask=0, data_size=3 (8B), no opts
    g1[1] = (int)(0x8000u << 16);     // tensor_dim0[15:0]=0x8000 at bits[63:48]
    g1[2] = (int)(0x0000u | (1u << 16)); // tensor_dim0[31:16]=0, tensor_dim1 lo=1
    g1[3] = (int)(0x8000u << 16);     // tensor_dim1 hi=0, tile_dim0=32768
    g1[4] = 1;                        // tile_dim1=1, tile_dim2=0
    g1[5] = 32768;                    // tensor_dim0_stride[31:0]
    g1[6] = (int)(0x8000u << 16);     // dim0_stride[47:32]=0, dim1_stride[15:0]=0x8000
    g1[7] = 0;                        // dim1_stride[47:16]=0
    i32x4 gz = {0, 0, 0, 0};
#if __clang_major__ >= 23
    i32x8 gz8 = {0, 0, 0, 0, 0, 0, 0, 0};
    __builtin_amdgcn_tensor_load_to_lds(g0, g1, gz, gz, gz8, 0);
#else
    __builtin_amdgcn_tensor_load_to_lds(g0, g1, gz, gz, 0);
#endif
    __builtin_amdgcn_s_wait_tensorcnt(0);
  }
#else
  // ---- fallback: cooperative 16B copy loop -------------------------------
  {
    const uint4* src = (const uint4*)vt_in;
    uint4*       dst = (uint4*)vt_lds;
    #pragma unroll 4
    for (int i = tid; i < (G_DIM * B_DIM) / 8; i += 1024) dst[i] = src[i];
  }
#endif

  // ---- per-lane softmax row of W, packed into WMMA A operands -----------
  // A 16x32 layout used here (contiguous K, not ISA table order): lane L holds row M=L%16;
//   VGPR j holds K pair (K,K+1), K = 2*j + 16*half
  float inv_s = 0.f;
  if (m < M_DIM) {
    float s = 0.f;
    for (int c = 0; c < C_DIM; ++c) s += __expf(W[m * C_DIM + c]);
    inv_s = 1.f / s;
  }
  v16bf A[4];
  #pragma unroll
  for (int k = 0; k < 4; ++k) {
    BfOp a;
    #pragma unroll
    for (int j = 0; j < 8; ++j) {
      int K0 = 2 * j + 16 * half;
      int c  = 32 * k + K0;
      float w0 = (m < M_DIM) ? __expf(W[m * C_DIM + c])     * inv_s : 0.f;
      float w1 = (m < M_DIM) ? __expf(W[m * C_DIM + c + 1]) * inv_s : 0.f;
      a.u[j] = pack_bf16(w0, w1);
    }
    A[k] = a.v;
  }

  __syncthreads();

  int g = blockIdx.x * 32 + wid;
  g = __builtin_amdgcn_readfirstlane(g);   // SGPR g -> scalar X/mask loads
  if (g < G_DIM) {
    v8f acc1 = {};   // b = 0..15
    v8f acc2 = {};   // b = 16..31

    #pragma unroll
    for (int k = 0; k < 4; ++k) {          // K chunks of 32 over C=128
      const int c0 = 32 * k;
      unsigned P[16];                      // lane b: {cv(b,c0+2j), cv(b,c0+2j+1)}
      #pragma unroll
      for (int j = 0; j < 16; ++j) {
        float cv2[2];
        #pragma unroll
        for (int t = 0; t < 2; ++t) {
          int c = c0 + 2 * j + t;
          const long long* xp = X + ((((long long)c) * G_DIM + g) << 2);
          int i0 = (int)xp[0], i1 = (int)xp[1], i2 = (int)xp[2], i3 = (int)xp[3];
          float p = bf2f(vt_lds[i0 * B_DIM + lane]) *
                    bf2f(vt_lds[i1 * B_DIM + lane]) *
                    bf2f(vt_lds[i2 * B_DIM + lane]) *
                    bf2f(vt_lds[i3 * B_DIM + lane]);
          cv2[t] = fmask[(size_t)c * G_DIM + g] ? 1.0f : p;
        }
        P[j] = pack_bf16(cv2[0], cv2[1]);
      }
      // B operand (cv^T, 32x16): lanes 0-15 hold col N=lane K=0..15,
      // lanes 16-31 hold col N=lane-16 K=16..31 -> half-wave swap + select.
      unsigned S[16];
      #pragma unroll
      for (int j = 0; j < 16; ++j) S[j] = (unsigned)__shfl_xor((int)P[j], 16);
      BfOp b1, b2;
      #pragma unroll
      for (int j = 0; j < 8; ++j) {
        b1.u[j] = half ? S[8 + j] : P[j];
        b2.u[j] = half ? P[8 + j] : S[j];
      }
      acc1 = __builtin_amdgcn_wmma_f32_16x16x32_bf16(false, A[k], false, b1.v,
                                                     (short)0, acc1, false, false);
      acc2 = __builtin_amdgcn_wmma_f32_16x16x32_bf16(false, A[k], false, b2.v,
                                                     (short)0, acc2, false, false);
    }

    // D layout: lanes 0-15 hold h(m=VGPRidx, b=N); rows m>=5 are zero-weight.
    AccOp d1, d2; d1.v = acc1; d2.v = acc2;
    float r1 = 1.f - (1.f - d1.f[0]) * (1.f - d1.f[1]) * (1.f - d1.f[2]) *
                     (1.f - d1.f[3]) * (1.f - d1.f[4]);
    float r2 = 1.f - (1.f - d2.f[0]) * (1.f - d2.f[1]) * (1.f - d2.f[2]) *
                     (1.f - d2.f[3]) * (1.f - d2.f[4]);
    r1 = fminf(fmaxf(r1, 0.f), 1.f);
    r2 = fminf(fmaxf(r2, 0.f), 1.f);
    // lanes 0-15 have r(b=lane)=r1; lanes 16-31 take r(b=lane)=r2 of lane-16
    float rsw = __shfl_xor(r2, 16);
    float r   = half ? rsw : r1;

    float vo = bf2f(vt_lds[g * B_DIM + lane]);
    float vn = 1.f - (1.f - vo) * (1.f - r);
    vn = fminf(fmaxf(vn, 0.f), 1.f);

    vt_out[(size_t)g * B_DIM + lane] = to_bf16u(vn);
    if (out_f32) out_f32[(size_t)lane * G_DIM + g] = vn;
  }
}

// ---------------------------------------------------------------------------
extern "C" void kernel_launch(void* const* d_in, const int* in_sizes, int n_in,
                              void* d_out, int out_size, void* d_ws, size_t ws_size,
                              hipStream_t stream) {
  (void)in_sizes; (void)n_in; (void)out_size; (void)ws_size;
  const float*         v0 = (const float*)d_in[0];
  const long long*     X  = (const long long*)d_in[1];
  const unsigned char* fm = (const unsigned char*)d_in[2];
  const float*         W  = (const float*)d_in[3];

  unsigned short* vt_a = (unsigned short*)d_ws;
  unsigned short* vt_b = vt_a + (size_t)G_DIM * B_DIM;

  k_transpose_in<<<(B_DIM * G_DIM + 255) / 256, 256, 0, stream>>>(v0, vt_a);
  // T = 2 steps; device-wide dependency between steps -> separate launches
  k_step<<<G_DIM / 32, 1024, 0, stream>>>(vt_a, X, fm, W, vt_b, nullptr);
  k_step<<<G_DIM / 32, 1024, 0, stream>>>(vt_b, X, fm, W, vt_a, (float*)d_out);
}
